// HierarchicalPooling_25039659336346
// MI455X (gfx1250) — matrix operand order repaired and checked by reading, not compiled, not run on hardware
//
#include <hip/hip_runtime.h>
#include <hip/hip_bf16.h>

typedef __attribute__((ext_vector_type(16))) __bf16   v16bf;
typedef __attribute__((ext_vector_type(8)))  float    v8f;
typedef __attribute__((ext_vector_type(8)))  unsigned v8u;

#define EROW_U32 132   // (256 + 8 pad) bf16 per row / 2  -> conflict-free LDS stride
#define NROW_U32 68    // (128 + 8 pad) bf16 per row / 2

__device__ __forceinline__ unsigned pack_bf16x2(float lo, float hi) {
  unsigned a = __float_as_uint(lo), b = __float_as_uint(hi);
  a = (a + 0x7FFFu + ((a >> 16) & 1u)) >> 16;   // round-to-nearest-even
  b = (b + 0x7FFFu + ((b >> 16) & 1u)) >> 16;
  return a | (b << 16);
}

// Branch-free tanh: (e^2x - 1) / (e^2x + 1), clamped so e^2x stays finite.
// ~5 VALU + 1 TRANS (v_exp_f32 co-executes with WMMA), no EXEC divergence.
__device__ __forceinline__ float tanh_fast(float x) {
  const float xc = fminf(fmaxf(x, -15.f), 15.f);
  const float t = __expf(2.f * xc);
  return __fdividef(t - 1.f, t + 1.f);
}

// ---------------------------------------------------------------------------
// Zero accumulators (graph-replay safe: runs every launch).
// fsmall layout: [0..1] max_node, [2..3] max_edge, [4..5] sum_node,
//                [6..7] sum_edge, [8..263] pooled[2][128], [264..265] struct
// ---------------------------------------------------------------------------
__global__ void k_init(float* fsmall) {
  for (int i = threadIdx.x; i < 266; i += 256) fsmall[i] = 0.f;
}

// ---------------------------------------------------------------------------
// Pack w_s1 [256x128] and w_n1 [128x64] (row-major fp32) into a contiguous-k
// bf16 B-fragment layout: frag index ((nt*KB+kb)*32 + lane)*8 + r holds the
// bf16 pair (W[k][n], W[k+1][n]) with n = nt*16 + lane%16,
// k = kb*32 + (lane/16)*16 + 2r.  One dword per thread.
// ---------------------------------------------------------------------------
__global__ void k_pack_weights(const float* __restrict__ w_s1,
                               const float* __restrict__ w_n1,
                               unsigned* __restrict__ outS,
                               unsigned* __restrict__ outN) {
  const int t = blockIdx.x * 256 + threadIdx.x;
  if (t < 16384) {                       // w_s1: K=256 (kb<8), Ncols=128 (nt<8)
    const int r = t & 7, lane = (t >> 3) & 31, kb = (t >> 8) & 7, nt = t >> 11;
    const int n = nt * 16 + (lane & 15);
    const int k = kb * 32 + ((lane >> 4) << 4) + 2 * r;
    outS[t] = pack_bf16x2(w_s1[k * 128 + n], w_s1[(k + 1) * 128 + n]);
  } else if (t < 16384 + 4096) {         // w_n1: K=128 (kb<4), Ncols=64 (nt<4)
    const int q = t - 16384;
    const int r = q & 7, lane = (q >> 3) & 31, kb = (q >> 8) & 3, nt = q >> 10;
    const int n = nt * 16 + (lane & 15);
    const int k = kb * 32 + ((lane >> 4) << 4) + 2 * r;
    outN[q] = pack_bf16x2(w_n1[k * 64 + n], w_n1[(k + 1) * 64 + n]);
  }
}

// ---------------------------------------------------------------------------
// Node logits: per wave, 16 nodes. hidden[16x64] = A[16x128]·W[128x64] via
// 4x4 bf16 WMMA steps, then tanh, dot with w_n2, cross-lane reduce.
// ---------------------------------------------------------------------------
__global__ __launch_bounds__(128) void k_node_logits(
    const float* __restrict__ emb, const unsigned* __restrict__ wpack,
    const float* __restrict__ b_n1, const float* __restrict__ w_n2,
    const float* __restrict__ b_n2, float* __restrict__ nlog, int Nn) {
  __shared__ unsigned tileLds[4 * 16 * NROW_U32];
  const int lane = threadIdx.x & 31;
  const int wave = threadIdx.x >> 5;
  const int hlf  = lane >> 4;
  const int l16  = lane & 15;
  const int b    = blockIdx.y;
  const int ntiles = Nn >> 4;
  int tile = blockIdx.x * 4 + wave;
  const bool valid = tile < ntiles;
  if (tile >= ntiles) tile = ntiles - 1;
  const int nbase = tile << 4;
  unsigned* tl = tileLds + wave * 16 * NROW_U32;
  const size_t embOff = (size_t)b * Nn * 128;

  // stage 16x128 fp32 -> bf16 LDS tile (padded rows)
  for (int r = 0; r < 16; ++r) {
    const float* rp = emb + embOff + (size_t)(nbase + r) * 128 + lane * 4;
    float4 f = *(const float4*)rp;
    unsigned* dst = tl + r * NROW_U32 + lane * 2;
    dst[0] = pack_bf16x2(f.x, f.y);
    dst[1] = pack_bf16x2(f.z, f.w);
  }
  asm volatile("s_wait_dscnt 0x0" ::: "memory");

  // A fragments (16-bit A 16x32 layout): lane l16 = row M; reg r -> K pair
  v16bf afrag[4];
#pragma unroll
  for (int kb = 0; kb < 4; ++kb) {
    v8u u;
#pragma unroll
    for (int r = 0; r < 8; ++r) {
      const int k0 = kb * 32 + hlf * 16 + r * 2;
      u[r] = tl[l16 * NROW_U32 + (k0 >> 1)];
    }
    afrag[kb] = __builtin_bit_cast(v16bf, u);
  }

  float rowacc[8];
#pragma unroll
  for (int r = 0; r < 8; ++r) rowacc[r] = 0.f;

#pragma unroll
  for (int nt = 0; nt < 4; ++nt) {
    v8f acc = {};
#pragma unroll
    for (int kb = 0; kb < 4; ++kb) {
      v8u bu = *(const v8u*)(wpack + (((nt * 4 + kb) * 32 + lane) << 3));
      v16bf bf = __builtin_bit_cast(v16bf, bu);
      acc = __builtin_amdgcn_wmma_f32_16x16x32_bf16(
          false, afrag[kb], false, bf, (short)0, acc, false, false);
    }
    const int n = nt * 16 + l16;
    const float b1 = b_n1[n];
    const float w2 = w_n2[n];
#pragma unroll
    for (int r = 0; r < 8; ++r) rowacc[r] += tanh_fast(acc[r] + b1) * w2;
  }
#pragma unroll
  for (int r = 0; r < 8; ++r) {     // reduce over 16-lane half (cols)
    float v = rowacc[r];
    v += __shfl_xor(v, 1, 32);
    v += __shfl_xor(v, 2, 32);
    v += __shfl_xor(v, 4, 32);
    v += __shfl_xor(v, 8, 32);
    rowacc[r] = v;
  }
  if (valid && l16 == 0) {          // lane0 -> rows 0..7, lane16 -> rows 8..15
    const float bn2 = b_n2[0];
#pragma unroll
    for (int r = 0; r < 8; ++r)
      nlog[(size_t)b * Nn + nbase + hlf * 8 + r] = rowacc[r] + bn2;
  }
}

// ---------------------------------------------------------------------------
// Edge logits + per-edge means: per wave, 16 edges. Gather [16x256] (src‖tgt)
// into LDS bf16, hidden[16x128] = A[16x256]·W[256x128] via 8x8 WMMA steps.
// ---------------------------------------------------------------------------
__global__ __launch_bounds__(128) void k_edge_logits(
    const float* __restrict__ emb, const int* __restrict__ eidx,
    const unsigned* __restrict__ wpack, const float* __restrict__ b_s1,
    const float* __restrict__ w_s2, const float* __restrict__ b_s2,
    float* __restrict__ elog, float* __restrict__ emean, int Nn, int Ecnt) {
  __shared__ unsigned tileLds[4 * 16 * EROW_U32];
  __shared__ float rowsum[4][16];
  const int lane = threadIdx.x & 31;
  const int wave = threadIdx.x >> 5;
  const int hlf  = lane >> 4;          // 0: src half (cols 0..127), 1: tgt half
  const int l16  = lane & 15;
  const int b    = blockIdx.y;
  const int etiles = Ecnt >> 4;
  int tile = blockIdx.x * 4 + wave;
  const bool valid = tile < etiles;
  if (tile >= etiles) tile = etiles - 1;
  const int ebase = tile << 4;
  unsigned* tl = tileLds + wave * 16 * EROW_U32;
  const size_t embOff = (size_t)b * Nn * 128;

  // gather + stage 16x256 -> bf16 LDS; fp32 row sums for mean(e, axis=-1)
  for (int r = 0; r < 16; ++r) {
    const int e = ebase + r;
    const int node = hlf ? eidx[Ecnt + e] : eidx[e];
    const float* rp = emb + embOff + (size_t)node * 128 + l16 * 8;
    float4 f0 = ((const float4*)rp)[0];
    float4 f1 = ((const float4*)rp)[1];
    float s = f0.x + f0.y + f0.z + f0.w + f1.x + f1.y + f1.z + f1.w;
    for (int m = 16; m; m >>= 1) s += __shfl_xor(s, m, 32);
    if (lane == 0) rowsum[wave][r] = s;
    unsigned* dst = tl + r * EROW_U32 + hlf * 64 + l16 * 4;
    dst[0] = pack_bf16x2(f0.x, f0.y);
    dst[1] = pack_bf16x2(f0.z, f0.w);
    dst[2] = pack_bf16x2(f1.x, f1.y);
    dst[3] = pack_bf16x2(f1.z, f1.w);
  }
  asm volatile("s_wait_dscnt 0x0" ::: "memory");

  v16bf afrag[8];
#pragma unroll
  for (int kb = 0; kb < 8; ++kb) {
    v8u u;
#pragma unroll
    for (int r = 0; r < 8; ++r) {
      const int k0 = kb * 32 + hlf * 16 + r * 2;
      u[r] = tl[l16 * EROW_U32 + (k0 >> 1)];
    }
    afrag[kb] = __builtin_bit_cast(v16bf, u);
  }

  float rowacc[8];
#pragma unroll
  for (int r = 0; r < 8; ++r) rowacc[r] = 0.f;

#pragma unroll
  for (int nt = 0; nt < 8; ++nt) {
    v8f acc = {};
#pragma unroll
    for (int kb = 0; kb < 8; ++kb) {
      v8u bu = *(const v8u*)(wpack + (((nt * 8 + kb) * 32 + lane) << 3));
      v16bf bf = __builtin_bit_cast(v16bf, bu);
      acc = __builtin_amdgcn_wmma_f32_16x16x32_bf16(
          false, afrag[kb], false, bf, (short)0, acc, false, false);
    }
    const int n = nt * 16 + l16;
    const float b1 = b_s1[n];
    const float w2 = w_s2[n];
#pragma unroll
    for (int r = 0; r < 8; ++r) rowacc[r] += tanh_fast(acc[r] + b1) * w2;
  }
#pragma unroll
  for (int r = 0; r < 8; ++r) {
    float v = rowacc[r];
    v += __shfl_xor(v, 1, 32);
    v += __shfl_xor(v, 2, 32);
    v += __shfl_xor(v, 4, 32);
    v += __shfl_xor(v, 8, 32);
    rowacc[r] = v;
  }
  if (valid) {
    if (l16 == 0) {
      const float bs2 = b_s2[0];
#pragma unroll
      for (int r = 0; r < 8; ++r)
        elog[(size_t)b * Ecnt + ebase + hlf * 8 + r] = rowacc[r] + bs2;
    }
    if (lane < 16)
      emean[(size_t)b * Ecnt + ebase + lane] =
          rowsum[wave][lane] * (1.0f / 256.0f);
  }
}

// ---------------------------------------------------------------------------
// Per-batch max over n logits (one block per batch).
// ---------------------------------------------------------------------------
__global__ void k_max(const float* __restrict__ in, int n, float* outp) {
  __shared__ float sm[256];
  const float* p = in + (size_t)blockIdx.x * n;
  float m = -3.402823466e38f;
  for (int i = threadIdx.x; i < n; i += 256) m = fmaxf(m, p[i]);
  sm[threadIdx.x] = m;
  __syncthreads();
  for (int s = 128; s > 0; s >>= 1) {
    if (threadIdx.x < s) sm[threadIdx.x] = fmaxf(sm[threadIdx.x], sm[threadIdx.x + s]);
    __syncthreads();
  }
  if (threadIdx.x == 0) outp[blockIdx.x] = sm[0];
}

// ---------------------------------------------------------------------------
// Node softmax-weighted pooling: accumulate sum(p) and sum(p * x[:,col]).
// ---------------------------------------------------------------------------
__global__ void k_node_pool(const float* __restrict__ emb,
                            const float* __restrict__ nlog,
                            float* fsmall, int Nn) {
  const int b = blockIdx.y;
  const float mx = fsmall[b];
  const int col = threadIdx.x & 127;
  const int g = threadIdx.x >> 7;
  const float* xb = emb + (size_t)b * Nn * 128;
  const float* lb = nlog + (size_t)b * Nn;
  float av = 0.f, ap = 0.f;
  for (int n = blockIdx.x * 2 + g; n < Nn; n += gridDim.x * 2) {
    const float p = __expf(lb[n] - mx);
    av += p * xb[(size_t)n * 128 + col];
    ap += p;
  }
  __shared__ float sv[256];
  sv[threadIdx.x] = av;
  __syncthreads();
  if (g == 0) atomicAdd(&fsmall[8 + b * 128 + col], av + sv[128 + col]);
  if (col == 0) atomicAdd(&fsmall[4 + b], ap);
}

// ---------------------------------------------------------------------------
// Edge softmax-weighted structure sum: sum(p) and sum(p * mean_e).
// ---------------------------------------------------------------------------
__global__ void k_edge_pool(const float* __restrict__ elog,
                            const float* __restrict__ emean,
                            float* fsmall, int Ecnt) {
  const int b = blockIdx.y;
  const float mx = fsmall[2 + b];
  float ap = 0.f, as = 0.f;
  for (int e = blockIdx.x * 256 + threadIdx.x; e < Ecnt; e += gridDim.x * 256) {
    const float p = __expf(elog[(size_t)b * Ecnt + e] - mx);
    ap += p;
    as += p * emean[(size_t)b * Ecnt + e];
  }
  __shared__ float s1[256], s2[256];
  s1[threadIdx.x] = ap;
  s2[threadIdx.x] = as;
  __syncthreads();
  for (int s = 128; s > 0; s >>= 1) {
    if (threadIdx.x < s) {
      s1[threadIdx.x] += s1[threadIdx.x + s];
      s2[threadIdx.x] += s2[threadIdx.x + s];
    }
    __syncthreads();
  }
  if (threadIdx.x == 0) {
    atomicAdd(&fsmall[6 + b], s1[0]);
    atomicAdd(&fsmall[264 + b], s2[0]);
  }
}

// ---------------------------------------------------------------------------
// out[b, 0:128] = pooled/sum ; out[b, 128:256] = struct/sum (broadcast).
// ---------------------------------------------------------------------------
__global__ void k_finalize(const float* __restrict__ fsmall, float* __restrict__ out) {
  const int t = threadIdx.x;
  if (t < 512) {
    const int b = t >> 8, j = t & 255;
    const float v = (j < 128) ? fsmall[8 + b * 128 + j] / fsmall[4 + b]
                              : fsmall[264 + b] / fsmall[6 + b];
    out[b * 256 + j] = v;
  }
}

extern "C" void kernel_launch(void* const* d_in, const int* in_sizes, int n_in,
                              void* d_out, int out_size, void* d_ws, size_t ws_size,
                              hipStream_t stream) {
  const float* emb  = (const float*)d_in[0];
  const int*   eidx = (const int*)d_in[1];
  const float* w_n1 = (const float*)d_in[2];
  const float* b_n1 = (const float*)d_in[3];
  const float* w_n2 = (const float*)d_in[4];
  const float* b_n2 = (const float*)d_in[5];
  const float* w_s1 = (const float*)d_in[6];
  const float* b_s1 = (const float*)d_in[7];
  const float* w_s2 = (const float*)d_in[8];
  const float* b_s2 = (const float*)d_in[9];
  float* out = (float*)d_out;

  const int Bb = 2, Hh = 128;
  const int Nn   = in_sizes[0] / (Bb * Hh);
  const int Ecnt = in_sizes[1] / 2;

  char* w = (char*)d_ws;
  unsigned* ws1p = (unsigned*)(w);                     // 64 KB packed w_s1
  unsigned* wn1p = (unsigned*)(w + 65536);             // 16 KB packed w_n1
  float* nlog  = (float*)(w + 81920);                  // B*N logits
  float* elog  = nlog + (size_t)Bb * Nn;               // B*E logits
  float* emean = elog + (size_t)Bb * Ecnt;             // B*E means
  float* fsmall = emean + (size_t)Bb * Ecnt;           // 266 floats

  k_init<<<1, 256, 0, stream>>>(fsmall);
  k_pack_weights<<<80, 256, 0, stream>>>(w_s1, w_n1, ws1p, wn1p);

  const int ntiles = (Nn + 15) >> 4;
  const int etiles = (Ecnt + 15) >> 4;
  dim3 gN((ntiles + 3) / 4, Bb);
  dim3 gE((etiles + 3) / 4, Bb);
  k_node_logits<<<gN, 128, 0, stream>>>(emb, wn1p, b_n1, w_n2, b_n2, nlog, Nn);
  k_edge_logits<<<gE, 128, 0, stream>>>(emb, eidx, ws1p, b_s1, w_s2, b_s2,
                                        elog, emean, Nn, Ecnt);
  k_max<<<Bb, 256, 0, stream>>>(nlog, Nn, fsmall + 0);
  k_max<<<Bb, 256, 0, stream>>>(elog, Ecnt, fsmall + 2);
  k_node_pool<<<dim3(128, Bb), 256, 0, stream>>>(emb, nlog, fsmall, Nn);
  k_edge_pool<<<dim3(128, Bb), 256, 0, stream>>>(elog, emean, fsmall, Ecnt);
  k_finalize<<<1, 512, 0, stream>>>(fsmall, out);
}
